// DeformableTransformerEncoderLayer_56899726737444
// MI455X (gfx1250) — matrix operand order repaired and checked by reading, not compiled, not run on hardware
//
#include <hip/hip_runtime.h>
#include <hip/hip_bf16.h>

// ---------------------------------------------------------------------------
// Problem constants (match reference)
// ---------------------------------------------------------------------------
#define DD    256
#define NH    8
#define NL    4
#define NP    4
#define HD    32          // DD / NH
#define DFF   1024
#define BB    2
#define LL    12240       // 96*96 + 48*48 + 24*24 + 12*12
#define MM    (BB * LL)   // 24480 token rows

typedef __attribute__((ext_vector_type(16))) __bf16 v16bf;
typedef __attribute__((ext_vector_type(8)))  __bf16 v8bf;
typedef __attribute__((ext_vector_type(8)))  float  v8f;

// ---------------------------------------------------------------------------
// Weight transpose + f32 -> bf16 convert:  Wt[n*K + k] = (bf16) W[k*N + n]
// ---------------------------------------------------------------------------
__global__ __launch_bounds__(256) void transpose_bf16_kernel(
    const float* __restrict__ W, __bf16* __restrict__ Wt, int K, int N)
{
    int i = blockIdx.x * 256 + threadIdx.x;
    if (i >= K * N) return;
    int k = i / N;
    int n = i - k * N;
    Wt[(size_t)n * K + k] = (__bf16)W[i];
}

// ---------------------------------------------------------------------------
// Elementwise add (q = src + pos), vectorized float4
// ---------------------------------------------------------------------------
__global__ __launch_bounds__(256) void add_kernel(
    const float* __restrict__ a, const float* __restrict__ b,
    float* __restrict__ c, int n4)
{
    int i = blockIdx.x * 256 + threadIdx.x;
    if (i >= n4) return;
    float4 x = ((const float4*)a)[i];
    float4 y = ((const float4*)b)[i];
    float4 z; z.x = x.x + y.x; z.y = x.y + y.y; z.z = x.z + y.z; z.w = x.w + y.w;
    ((float4*)c)[i] = z;
}

// ---------------------------------------------------------------------------
// Generic bf16 WMMA GEMM:  C[M,N] = A[M,K] @ Wt^T + bias   (Wt is N x K, bf16)
//   - 128x64 output tile per 256-thread block (8 wave32s in a 4x2 grid)
//   - each wave: 32x32 tile = four v_wmma_f32_16x16x32_bf16 accumulators
//   - weight (and bf16-A) tiles staged via GLOBAL_LOAD_ASYNC_TO_LDS_B128
//     (ASYNCcnt path); f32 A staged with VALU cvt + ds_store (needs convert)
// ---------------------------------------------------------------------------
template <bool A_BF16, bool RELU, bool STORE_BF16>
__global__ __launch_bounds__(256) void gemm_bf16_wmma(
    const void* __restrict__ Av, const __bf16* __restrict__ Wt,
    const float* __restrict__ bias, void* __restrict__ Cv,
    int M, int N, int K)
{
    __shared__ __align__(64) __bf16 lA[128][40]; // [row m][k], 80B padded rows
    __shared__ __align__(64) __bf16 lB[64][40];  // [row n][k]

    const int t    = threadIdx.x;
    const int lane = t & 31;
    const int w    = t >> 5;
    const int wm   = w >> 1;          // 0..3 -> 32-row slab
    const int wn   = w & 1;           // 0..1 -> 32-col slab
    const int bm   = blockIdx.y * 128;
    const int bn   = blockIdx.x * 64;

    // staging maps
    const int arow = t >> 1;          // A: 128 rows, 16 elems per thread
    const int acol = (t & 1) * 16;    //    cols 0 or 16
    const int brow = t >> 2;          // B: 64 rows, 8 elems per thread
    const int bcol = (t & 3) * 8;

    v8f c00 = {}, c01 = {}, c10 = {}, c11 = {};

    const int half  = lane >> 4;      // 16-lane group
    const int mrow0 = wm * 32 + (lane & 15);
    const int mrow1 = mrow0 + 16;
    const int nrow0 = wn * 32 + (lane & 15);
    const int nrow1 = nrow0 + 16;

    for (int k0 = 0; k0 < K; k0 += 32) {
        __syncthreads();   // LDS reuse guard

        // ---- stage B tile (64 n-rows x 32 k): async DMA, raw bf16 bytes ----
        {
            const __bf16* gsrc = Wt + (size_t)(bn + brow) * K + k0 + bcol;
            const unsigned ldsOff = (unsigned)(uintptr_t)&lB[brow][bcol];
            asm volatile("global_load_async_to_lds_b128 %0, %1, off"
                         :: "v"(ldsOff), "v"(gsrc) : "memory");
        }
        // ---- stage A tile (128 rows x 32 k) ----
        {
            const int gm = bm + arow;
            if constexpr (A_BF16) {
                if (gm < M) {
                    const __bf16* gsrc = (const __bf16*)Av + (size_t)gm * K + k0 + acol;
                    const unsigned o0 = (unsigned)(uintptr_t)&lA[arow][acol];
                    const unsigned o1 = (unsigned)(uintptr_t)&lA[arow][acol + 8];
                    asm volatile("global_load_async_to_lds_b128 %0, %1, off"
                                 :: "v"(o0), "v"(gsrc) : "memory");
                    asm volatile("global_load_async_to_lds_b128 %0, %1, off"
                                 :: "v"(o1), "v"(gsrc + 8) : "memory");
                } else {
                    #pragma unroll
                    for (int i = 0; i < 16; ++i) lA[arow][acol + i] = (__bf16)0.0f;
                }
            } else {
                __bf16 tmp[16];
                if (gm < M) {
                    const float4* s4 = (const float4*)((const float*)Av + (size_t)gm * K + k0 + acol);
                    float4 v0 = s4[0], v1 = s4[1], v2 = s4[2], v3 = s4[3];
                    float fv[16] = {v0.x, v0.y, v0.z, v0.w, v1.x, v1.y, v1.z, v1.w,
                                    v2.x, v2.y, v2.z, v2.w, v3.x, v3.y, v3.z, v3.w};
                    #pragma unroll
                    for (int i = 0; i < 16; ++i) tmp[i] = (__bf16)fv[i];
                } else {
                    #pragma unroll
                    for (int i = 0; i < 16; ++i) tmp[i] = (__bf16)0.0f;
                }
                #pragma unroll
                for (int i = 0; i < 16; ++i) lA[arow][acol + i] = tmp[i];
            }
        }
        // all async copies into LDS must land before the barrier releases readers
        asm volatile("s_wait_asynccnt 0x0" ::: "memory");
        __syncthreads();

        // ---- A fragments: e0..7 -> k=half*8+e, e8..15 -> k=16+half*8+e ----
        v8bf a00 = *(const v8bf*)&lA[mrow0][half * 8];
        v8bf a01 = *(const v8bf*)&lA[mrow0][16 + half * 8];
        v8bf a10 = *(const v8bf*)&lA[mrow1][half * 8];
        v8bf a11 = *(const v8bf*)&lA[mrow1][16 + half * 8];
        v16bf aF0, aF1;
        #pragma unroll
        for (int i = 0; i < 8; ++i) {
            aF0[i] = a00[i]; aF0[8 + i] = a01[i];
            aF1[i] = a10[i]; aF1[8 + i] = a11[i];
        }
        // ---- B fragments: lane = column, e0..7 -> k=half*8+e, e8..15 -> k=16+half*8+e ----
        v8bf b00 = *(const v8bf*)&lB[nrow0][half * 8];
        v8bf b01 = *(const v8bf*)&lB[nrow0][16 + half * 8];
        v8bf b10 = *(const v8bf*)&lB[nrow1][half * 8];
        v8bf b11 = *(const v8bf*)&lB[nrow1][16 + half * 8];
        v16bf bF0, bF1;
        #pragma unroll
        for (int i = 0; i < 8; ++i) {
            bF0[i] = b00[i]; bF0[8 + i] = b01[i];
            bF1[i] = b10[i]; bF1[8 + i] = b11[i];
        }

        c00 = __builtin_amdgcn_wmma_f32_16x16x32_bf16(false, aF0, false, bF0, (short)0, c00, false, false);
        c01 = __builtin_amdgcn_wmma_f32_16x16x32_bf16(false, aF0, false, bF1, (short)0, c01, false, false);
        c10 = __builtin_amdgcn_wmma_f32_16x16x32_bf16(false, aF1, false, bF0, (short)0, c10, false, false);
        c11 = __builtin_amdgcn_wmma_f32_16x16x32_bf16(false, aF1, false, bF1, (short)0, c11, false, false);
    }

    // ---- epilogue: C VGPR v -> m = base + (lane>=16 ? 8 : 0) + v, n = lane&15 ----
    const int mb0 = bm + wm * 32 + half * 8;
    const int mb1 = mb0 + 16;
    const int nc0 = bn + nrow0;
    const int nc1 = bn + nrow1;
    const float bias0 = bias[nc0];
    const float bias1 = bias[nc1];
    #pragma unroll
    for (int v = 0; v < 8; ++v) {
        const int m0 = mb0 + v;
        const int m1 = mb1 + v;
        float r00 = c00[v] + bias0, r01 = c01[v] + bias1;
        float r10 = c10[v] + bias0, r11 = c11[v] + bias1;
        if constexpr (RELU) {
            r00 = fmaxf(r00, 0.0f); r01 = fmaxf(r01, 0.0f);
            r10 = fmaxf(r10, 0.0f); r11 = fmaxf(r11, 0.0f);
        }
        if (m0 < M) {
            if constexpr (STORE_BF16) {
                ((__bf16*)Cv)[(size_t)m0 * N + nc0] = (__bf16)r00;
                ((__bf16*)Cv)[(size_t)m0 * N + nc1] = (__bf16)r01;
            } else {
                ((float*)Cv)[(size_t)m0 * N + nc0] = r00;
                ((float*)Cv)[(size_t)m0 * N + nc1] = r01;
            }
        }
        if (m1 < M) {
            if constexpr (STORE_BF16) {
                ((__bf16*)Cv)[(size_t)m1 * N + nc0] = (__bf16)r10;
                ((__bf16*)Cv)[(size_t)m1 * N + nc1] = (__bf16)r11;
            } else {
                ((float*)Cv)[(size_t)m1 * N + nc0] = r10;
                ((float*)Cv)[(size_t)m1 * N + nc1] = r11;
            }
        }
    }
}

// ---------------------------------------------------------------------------
// Softmax over 16 logits per (token, head). One thread per row, contiguous.
// ---------------------------------------------------------------------------
__global__ __launch_bounds__(256) void softmax16_kernel(float* __restrict__ p, int rows)
{
    int r = blockIdx.x * 256 + threadIdx.x;
    if (r >= rows) return;
    float* row = p + (size_t)r * 16;
    float mx = row[0];
    #pragma unroll
    for (int j = 1; j < 16; ++j) mx = fmaxf(mx, row[j]);
    float e[16]; float s = 0.0f;
    #pragma unroll
    for (int j = 0; j < 16; ++j) { e[j] = __expf(row[j] - mx); s += e[j]; }
    const float inv = 1.0f / s;
    #pragma unroll
    for (int j = 0; j < 16; ++j) row[j] = e[j] * inv;
}

// ---------------------------------------------------------------------------
// Deformable attention sampling core.
//   block = token (B*L blocks), wave = head, lane = channel (HD == 32).
//   value is (B*L, 256) with column = h*32 + c; gathers are 128B/corner, L2-hot.
// ---------------------------------------------------------------------------
__global__ __launch_bounds__(256) void deform_attn_kernel(
    const float* __restrict__ value,   // (B*L, 256)
    const float* __restrict__ off,     // (B*L, 256)
    const float* __restrict__ aw,      // (B*L, 128), post-softmax
    const float* __restrict__ refp,    // (B, L, NL, 2)
    float* __restrict__ out)           // (B*L, 256)
{
    const int SZ[4]     = {96, 48, 24, 12};
    const int STARTS[4] = {0, 9216, 11520, 12096};

    const int tok = blockIdx.x;          // 0 .. B*L-1
    const int h   = threadIdx.x >> 5;    // head
    const int c   = threadIdx.x & 31;    // channel
    const int b   = tok / LL;

    const float* offrow = off  + (size_t)tok * 256 + h * 32;
    const float* awrow  = aw   + (size_t)tok * 128 + h * 16;
    const float* refrow = refp + (size_t)tok * (NL * 2);
    const float* vbase  = value + (size_t)b * LL * 256 + h * 32 + c;

    float acc = 0.0f;
    #pragma unroll
    for (int l = 0; l < 4; ++l) {
        const int   Wl = SZ[l];
        const float rx = refrow[l * 2 + 0];
        const float ry = refrow[l * 2 + 1];
        const float* vl = vbase + (size_t)STARTS[l] * 256;
        #pragma unroll
        for (int p = 0; p < 4; ++p) {
            const int j = l * 4 + p;
            const float ox  = offrow[j * 2 + 0];
            const float oy  = offrow[j * 2 + 1];
            const float wgt = awrow[j];
            // loc*W - 0.5 with loc = ref + off/W  ==>  ref*W + off - 0.5
            const float x = rx * (float)Wl + ox - 0.5f;
            const float y = ry * (float)Wl + oy - 0.5f;
            const float x0f = floorf(x), y0f = floorf(y);
            const int   x0  = (int)x0f,  y0  = (int)y0f;
            const float wx  = x - x0f,   wy  = y - y0f;

            float s = 0.0f;
            #pragma unroll
            for (int dy = 0; dy < 2; ++dy) {
                #pragma unroll
                for (int dx = 0; dx < 2; ++dx) {
                    const int ix = x0 + dx, iy = y0 + dy;
                    const bool valid = ((unsigned)ix < (unsigned)Wl) &
                                       ((unsigned)iy < (unsigned)Wl);
                    const int cx = ix < 0 ? 0 : (ix >= Wl ? Wl - 1 : ix);
                    const int cy = iy < 0 ? 0 : (iy >= Wl ? Wl - 1 : iy);
                    const float v  = vl[(size_t)(cy * Wl + cx) * 256];
                    const float ww = (dx ? wx : 1.0f - wx) * (dy ? wy : 1.0f - wy);
                    s += valid ? v * ww : 0.0f;
                }
            }
            acc += wgt * s;
        }
    }
    out[(size_t)tok * 256 + h * 32 + c] = acc;
}

// ---------------------------------------------------------------------------
// Fused residual + LayerNorm over D=256. One wave per token, shfl reductions.
// ---------------------------------------------------------------------------
__global__ __launch_bounds__(256) void residual_ln_kernel(
    const float* __restrict__ x, const float* __restrict__ r,
    const float* __restrict__ g, const float* __restrict__ bta,
    float* __restrict__ out, int M)
{
    const int tok = blockIdx.x * 8 + (threadIdx.x >> 5);
    if (tok >= M) return;
    const int lane = threadIdx.x & 31;
    const float* xr = x + (size_t)tok * 256;
    const float* rr = r + (size_t)tok * 256;

    float v[8]; float s = 0.0f, s2 = 0.0f;
    #pragma unroll
    for (int i = 0; i < 8; ++i) {
        const int cidx = lane + i * 32;
        const float val = xr[cidx] + rr[cidx];
        v[i] = val; s += val; s2 += val * val;
    }
    #pragma unroll
    for (int m = 16; m > 0; m >>= 1) {
        s  += __shfl_xor(s,  m, 32);
        s2 += __shfl_xor(s2, m, 32);
    }
    const float mean = s * (1.0f / 256.0f);
    const float var  = s2 * (1.0f / 256.0f) - mean * mean;
    const float inv  = rsqrtf(var + 1e-5f);
    float* orow = out + (size_t)tok * 256;
    #pragma unroll
    for (int i = 0; i < 8; ++i) {
        const int cidx = lane + i * 32;
        orow[cidx] = (v[i] - mean) * inv * g[cidx] + bta[cidx];
    }
}

// ---------------------------------------------------------------------------
// Host-side orchestration
// ---------------------------------------------------------------------------
extern "C" void kernel_launch(void* const* d_in, const int* in_sizes, int n_in,
                              void* d_out, int out_size, void* d_ws, size_t ws_size,
                              hipStream_t stream)
{
    (void)in_sizes; (void)n_in; (void)out_size; (void)ws_size;

    const float* src     = (const float*)d_in[0];
    const float* pos     = (const float*)d_in[1];
    const float* refp    = (const float*)d_in[2];
    const float* value_w = (const float*)d_in[3];
    const float* value_b = (const float*)d_in[4];
    const float* off_w   = (const float*)d_in[5];
    const float* off_b   = (const float*)d_in[6];
    const float* attn_w  = (const float*)d_in[7];
    const float* attn_b  = (const float*)d_in[8];
    const float* out_w   = (const float*)d_in[9];
    const float* out_b   = (const float*)d_in[10];
    const float* ln1_g   = (const float*)d_in[11];
    const float* ln1_b   = (const float*)d_in[12];
    const float* lin1_w  = (const float*)d_in[13];
    const float* lin1_b  = (const float*)d_in[14];
    const float* lin2_w  = (const float*)d_in[15];
    const float* lin2_b  = (const float*)d_in[16];
    const float* ln3_g   = (const float*)d_in[17];
    const float* ln3_b   = (const float*)d_in[18];
    // d_in[19] spatial_shapes (hard-coded), d_in[20] padding_mask (all false)

    // ---- workspace carve-up (256B aligned) ----
    char* ws = (char*)d_ws;
    size_t o = 0;
    auto carve = [&](size_t bytes) -> char* {
        char* p = ws + o;
        o = (o + bytes + 255) & ~(size_t)255;
        return p;
    };
    __bf16* wT_value = (__bf16*)carve((size_t)256 * 256 * 2);
    __bf16* wT_off   = (__bf16*)carve((size_t)256 * 256 * 2);
    __bf16* wT_attn  = (__bf16*)carve((size_t)128 * 256 * 2);
    __bf16* wT_out   = (__bf16*)carve((size_t)256 * 256 * 2);
    __bf16* wT_lin1  = (__bf16*)carve((size_t)1024 * 256 * 2);
    __bf16* wT_lin2  = (__bf16*)carve((size_t)256 * 1024 * 2);
    float*  bufA = (float*)carve((size_t)MM * 256 * 4);  // q        -> attn_in
    float*  bufB = (float*)carve((size_t)MM * 256 * 4);  // value    -> src1
    float*  bufC = (float*)carve((size_t)MM * 256 * 4);  // off      -> attn_out -> ffn
    float*  bufD = (float*)carve((size_t)MM * 128 * 4);  // logits   -> aw
    __bf16* bufH = (__bf16*)carve((size_t)MM * DFF * 2); // FFN hidden (bf16)

    const dim3 blk(256);

    // 1) weight transposes (f32 -> bf16, N-major)
    transpose_bf16_kernel<<<dim3((256 * 256 + 255) / 256), blk, 0, stream>>>(value_w, wT_value, 256, 256);
    transpose_bf16_kernel<<<dim3((256 * 256 + 255) / 256), blk, 0, stream>>>(off_w,   wT_off,   256, 256);
    transpose_bf16_kernel<<<dim3((256 * 128 + 255) / 256), blk, 0, stream>>>(attn_w,  wT_attn,  256, 128);
    transpose_bf16_kernel<<<dim3((256 * 256 + 255) / 256), blk, 0, stream>>>(out_w,   wT_out,   256, 256);
    transpose_bf16_kernel<<<dim3((256 * 1024 + 255) / 256), blk, 0, stream>>>(lin1_w, wT_lin1, 256, 1024);
    transpose_bf16_kernel<<<dim3((1024 * 256 + 255) / 256), blk, 0, stream>>>(lin2_w, wT_lin2, 1024, 256);

    // 2) q = src + pos
    const int n4 = MM * 256 / 4;
    add_kernel<<<dim3((n4 + 255) / 256), blk, 0, stream>>>(src, pos, bufA, n4);

    const int gy = (MM + 127) / 128;  // 192 M-tiles

    // 3) value = src @ value_w + b
    gemm_bf16_wmma<false, false, false><<<dim3(256 / 64, gy), blk, 0, stream>>>(src,  wT_value, value_b, bufB, MM, 256, 256);
    // 4) off = q @ off_w + b
    gemm_bf16_wmma<false, false, false><<<dim3(256 / 64, gy), blk, 0, stream>>>(bufA, wT_off,   off_b,   bufC, MM, 256, 256);
    // 5) logits = q @ attn_w + b
    gemm_bf16_wmma<false, false, false><<<dim3(128 / 64, gy), blk, 0, stream>>>(bufA, wT_attn,  attn_b,  bufD, MM, 128, 256);

    // 6) softmax over 16 per (token, head)
    const int rows = MM * NH;
    softmax16_kernel<<<dim3((rows + 255) / 256), blk, 0, stream>>>(bufD, rows);

    // 7) deformable sampling -> attn_in (reuses bufA; q is dead)
    deform_attn_kernel<<<dim3(MM), blk, 0, stream>>>(bufB, bufC, bufD, refp, bufA);

    // 8) attn_out = attn_in @ out_w + b (reuses bufC)
    gemm_bf16_wmma<false, false, false><<<dim3(256 / 64, gy), blk, 0, stream>>>(bufA, wT_out, out_b, bufC, MM, 256, 256);

    // 9) src1 = LN(src + attn_out) -> bufB (value is dead)
    residual_ln_kernel<<<dim3((MM + 7) / 8), blk, 0, stream>>>(bufC, src, ln1_g, ln1_b, bufB, MM);

    // 10) hidden = relu(src1 @ lin1_w + b), stored bf16
    gemm_bf16_wmma<false, true, true><<<dim3(DFF / 64, gy), blk, 0, stream>>>(bufB, wT_lin1, lin1_b, bufH, MM, DFF, 256);

    // 11) ffn = hidden @ lin2_w + b (bf16 A path, K=1024) -> bufC
    gemm_bf16_wmma<true, false, false><<<dim3(256 / 64, gy), blk, 0, stream>>>(bufH, wT_lin2, lin2_b, bufC, MM, 256, DFF);

    // 12) out = LN(src1 + ffn)
    residual_ln_kernel<<<dim3((MM + 7) / 8), blk, 0, stream>>>(bufC, bufB, ln3_g, ln3_b, (float*)d_out, MM);
}
